// SoftGraphConditionedTwoWayBlock_15719580303534
// MI455X (gfx1250) — matrix operand order repaired and checked by reading, not compiled, not run on hardware
//
#include <hip/hip_runtime.h>
#include <stdint.h>

// ---------------------------------------------------------------------------
// Types for CDNA5 WMMA (gfx1250, wave32)
// ---------------------------------------------------------------------------
typedef __bf16 bf16_t;
typedef __attribute__((ext_vector_type(16))) __bf16 v16bf;
typedef __attribute__((ext_vector_type(8)))  float  v8f;

static __device__ __forceinline__ unsigned short f2bf_bits(float x) {
    union { float f; unsigned u; } a; a.f = x;
    unsigned r = a.u + 0x7FFFu + ((a.u >> 16) & 1u);   // round-to-nearest-even
    return (unsigned short)(r >> 16);
}
static __device__ __forceinline__ bf16_t f2bf(float x) {
    union { unsigned short s; bf16_t b; } o; o.s = f2bf_bits(x); return o.b;
}

// A-fragment (16x32 bf16): per-lane 16 contiguous elements
// (caller pre-offsets pointer by (lane>=16 ? 16 : 0) elements)
static __device__ __forceinline__ v16bf load_fragA(const bf16_t* p) {
    union { v16bf v; uint4 u[2]; } f;
    f.u[0] = *(const uint4*)(p);
    f.u[1] = *(const uint4*)(p + 8);
    return f.v;
}
// B-fragment (32x16 bf16): per-lane 16 contiguous elements
// (caller pre-offsets pointer by (lane>=16 ? 16 : 0) elements)
static __device__ __forceinline__ v16bf load_fragB(const bf16_t* p) {
    union { v16bf v; uint4 u[2]; } f;
    f.u[0] = *(const uint4*)(p);
    f.u[1] = *(const uint4*)(p + 8);
    return f.v;
}
static __device__ __forceinline__ v8f wmma_bf16(v16bf a, v16bf b, v8f c) {
    return __builtin_amdgcn_wmma_f32_16x16x32_bf16(false, a, false, b,
                                                   (short)0, c, false, false);
}

// ---------------------------------------------------------------------------
// Row remapping (fuses transposes / train-test splits into GEMM addressing)
// ---------------------------------------------------------------------------
struct MapP { int d1, d2; };
// 0: identity
// 1: train rows  : logical m in [0, blocks*192) -> phys (m/192)*256 + m%192
// 2: test rows   : logical m in [0, blocks*64)  -> phys (m/64)*256 + 192 + m%64
// 3: transpose   : r = (b*d1+i)*d2+j  -> (b*d2+j)*d1+i
template<int MAP>
static __device__ __forceinline__ int map_row(int r, MapP mp) {
    if (MAP == 1) { int blk = r / 192; return blk * 256 + (r - blk * 192); }
    if (MAP == 2) { int blk = r / 64;  return blk * 256 + 192 + (r - blk * 64); }
    if (MAP == 3) {
        int d12 = mp.d1 * mp.d2;
        int b = r / d12; int rem = r - b * d12;
        int i = rem / mp.d2; int j = rem - i * mp.d2;
        return (b * mp.d2 + j) * mp.d1 + i;
    }
    return r;
}

// ---------------------------------------------------------------------------
// f32 -> bf16 convert (weights)
// ---------------------------------------------------------------------------
__global__ void cvt_bf16_k(const float* __restrict__ in, bf16_t* __restrict__ out, int n) {
    int i = blockIdx.x * 256 + threadIdx.x;
    if (i < n) out[i] = f2bf(in[i]);
}

// ---------------------------------------------------------------------------
// LayerNorm over D=256, one wave per row, writes bf16
// ---------------------------------------------------------------------------
template<int MAP>
__global__ __launch_bounds__(256) void ln_k(const float* __restrict__ X,
                                            bf16_t* __restrict__ Y,
                                            const float* __restrict__ g,
                                            const float* __restrict__ b,
                                            int R, MapP mp) {
    int lane = threadIdx.x & 31;
    int row  = blockIdx.x * 8 + (threadIdx.x >> 5);
    if (row >= R) return;
    size_t pr = (size_t)map_row<MAP>(row, mp) * 256;
    const float* xp = X + pr + lane * 8;
    float4 a = *(const float4*)(xp);
    float4 c = *(const float4*)(xp + 4);
    float v[8] = {a.x, a.y, a.z, a.w, c.x, c.y, c.z, c.w};

    float s = 0.f;
#pragma unroll
    for (int i = 0; i < 8; i++) s += v[i];
#pragma unroll
    for (int m = 16; m >= 1; m >>= 1) s += __shfl_xor(s, m, 32);
    float mean = s * (1.0f / 256.0f);

    float q = 0.f;
#pragma unroll
    for (int i = 0; i < 8; i++) { float d = v[i] - mean; q += d * d; }
#pragma unroll
    for (int m = 16; m >= 1; m >>= 1) q += __shfl_xor(q, m, 32);
    float rstd = rsqrtf(q * (1.0f / 256.0f) + 1e-5f);

    union { uint4 u; unsigned short h[8]; } o;
#pragma unroll
    for (int i = 0; i < 8; i++) {
        int col = lane * 8 + i;
        o.h[i] = f2bf_bits((v[i] - mean) * rstd * g[col] + b[col]);
    }
    *(uint4*)(Y + pr + lane * 8) = o.u;
}

// ---------------------------------------------------------------------------
// Generic bf16 WMMA GEMM:  out = A[M,K] @ W[N,K]^T + bias  (+epilogue)
// 256 threads = 8 waves; each wave owns a 32x32 tile (2x2 WMMA frags).
// EPI: 0 = store bf16, 1 = +resid, store f32, 2 = GELU, store bf16
// ---------------------------------------------------------------------------
template<int INMAP, int OUTMAP, int EPI>
__global__ __launch_bounds__(256) void gemm_k(
    const bf16_t* __restrict__ A, int lda,
    const bf16_t* __restrict__ W, int K,
    const float* __restrict__ bias,
    const float* __restrict__ resid, int ldr,
    void* __restrict__ outp, int ldo,
    int M, int N, MapP mpin, MapP mpout)
{
    const int lane = threadIdx.x & 31;
    const int wid  = threadIdx.x >> 5;
    const int wm = wid & 3, wn = wid >> 2;
    const int half = lane >> 4, l16 = lane & 15;
    const int bm = blockIdx.y * 128 + wm * 32;
    const int bn = blockIdx.x * 64  + wn * 32;

    const bf16_t* ap[2];
    const bf16_t* wp[2];
#pragma unroll
    for (int i = 0; i < 2; i++) {
        int lrow = bm + i * 16 + l16;
        ap[i] = A + (size_t)map_row<INMAP>(lrow, mpin) * lda + half * 16;
        wp[i] = W + (size_t)(bn + i * 16 + l16) * K + half * 16;
    }

    const v8f vz = {0.f, 0.f, 0.f, 0.f, 0.f, 0.f, 0.f, 0.f};
    v8f acc[2][2] = {{vz, vz}, {vz, vz}};

    for (int k0 = 0; k0 < K; k0 += 32) {
        v16bf a0 = load_fragA(ap[0] + k0);
        v16bf a1 = load_fragA(ap[1] + k0);
        v16bf b0 = load_fragB(wp[0] + k0);
        v16bf b1 = load_fragB(wp[1] + k0);
        acc[0][0] = wmma_bf16(a0, b0, acc[0][0]);
        acc[0][1] = wmma_bf16(a0, b1, acc[0][1]);
        acc[1][0] = wmma_bf16(a1, b0, acc[1][0]);
        acc[1][1] = wmma_bf16(a1, b1, acc[1][1]);
    }

#pragma unroll
    for (int i = 0; i < 2; i++) {
#pragma unroll
        for (int j = 0; j < 2; j++) {
            int col = bn + j * 16 + l16;
            float bv = bias[col];
#pragma unroll
            for (int r = 0; r < 8; r++) {
                int rowL = bm + i * 16 + r + half * 8;
                float vv = acc[i][j][r] + bv;
                int orow = map_row<OUTMAP>(rowL, mpout);
                if (EPI == 1) {
                    int rrow = map_row<INMAP>(rowL, mpin);
                    vv += resid[(size_t)rrow * ldr + col];
                    ((float*)outp)[(size_t)orow * ldo + col] = vv;
                } else if (EPI == 2) {
                    float ge = 0.5f * vv * (1.0f + erff(vv * 0.70710678118654752f));
                    ((unsigned short*)outp)[(size_t)orow * ldo + col] = f2bf_bits(ge);
                } else {
                    ((unsigned short*)outp)[(size_t)orow * ldo + col] = f2bf_bits(vv);
                }
            }
        }
    }
}

// ---------------------------------------------------------------------------
// Attention: one block = (pair, 64-q-tile, head). 128 threads = 4 waves.
// scores = Q@K^T (WMMA from global), softmax (+ optional graph bias) in LDS,
// out = attn@V (WMMA from LDS with V transposed in LDS). dh = 32.
// ---------------------------------------------------------------------------
#define SP 200   // padded Skv stride (max Skv = 192)

template<bool HAS_BIAS>
__global__ __launch_bounds__(128) void attn_k(
    const bf16_t* __restrict__ Q, long long qps, int qrs,
    const bf16_t* __restrict__ K, long long kps, int krs,
    const bf16_t* __restrict__ V, long long vps, int vrs,
    bf16_t* __restrict__ O, long long ops, int ors,
    int nqt, int Skv,
    const float* __restrict__ graph,
    const float* __restrict__ gsc,
    const float* __restrict__ gof,
    int pairs_per_b)
{
    __shared__ float          s_sc[64 * SP];   // scores f32
    __shared__ unsigned short s_at[64 * SP];   // attn bf16
    __shared__ unsigned short s_vt[32 * SP];   // V^T bf16

    const int p  = blockIdx.x / nqt;
    const int qt = blockIdx.x % nqt;
    const int h  = blockIdx.y;
    const int tid = threadIdx.x, lane = tid & 31, w = tid >> 5;
    const int half = lane >> 4, l16 = lane & 15;

    const bf16_t* Qb = Q + (long long)p * qps + (long long)qt * 64 * qrs + h * 32;
    const bf16_t* Kb = K + (long long)p * kps + h * 32;
    const bf16_t* Vb = V + (long long)p * vps + h * 32;
    bf16_t*       Ob = O + (long long)p * ops + (long long)qt * 64 * ors + h * 32;

    // V^T into LDS
    for (int idx = tid; idx < Skv * 32; idx += 128) {
        int k = idx >> 5, n = idx & 31;
        s_vt[n * SP + k] = ((const unsigned short*)Vb)[(size_t)k * vrs + n];
    }

    // scores: each wave owns 16 q-rows, loops over all k-tiles
    v16bf aq = load_fragA(Qb + (size_t)(w * 16 + l16) * qrs + half * 16);
    for (int nt = 0; nt < (Skv >> 4); nt++) {
        v16bf bk = load_fragB(Kb + (size_t)(nt * 16 + l16) * krs + half * 16);
        v8f c = {0.f, 0.f, 0.f, 0.f, 0.f, 0.f, 0.f, 0.f};
        c = wmma_bf16(aq, bk, c);
#pragma unroll
        for (int r = 0; r < 8; r++)
            s_sc[(w * 16 + r + half * 8) * SP + nt * 16 + l16] = c[r];
    }
    __syncthreads();

    // softmax per row (with optional graph bias)
    if (tid < 64) {
        const int row = tid;
        const float rs = 0.17677669529663687f;   // 1/sqrt(32)
        float scale_h = 0.f, off_h = 0.f;
        const float* grow = nullptr;
        if (HAS_BIAS) {
            int b = p / pairs_per_b;
            int qf = qt * 64 + row;
            scale_h = gsc[h]; off_h = gof[h];
            grow = graph + ((size_t)b * 64 + qf) * 64;
        }
        float* srow = s_sc + row * SP;
        float mx = -1e30f;
        for (int c = 0; c < Skv; c++) {
            float v = srow[c] * rs;
            if (HAS_BIAS)
                v += scale_h * (grow[c] == 1.0f ? 0.0f : -5.0f) + off_h;
            srow[c] = v;
            mx = fmaxf(mx, v);
        }
        float s = 0.f;
        for (int c = 0; c < Skv; c++) { float e = __expf(srow[c] - mx); s += e; srow[c] = e; }
        float inv = 1.0f / s;
        unsigned short* arow = s_at + row * SP;
        for (int c = 0; c < Skv; c++) arow[c] = f2bf_bits(srow[c] * inv);
    }
    __syncthreads();

    // out = attn @ V   (contraction over Skv in 32-chunks)
    v8f o0 = {0.f, 0.f, 0.f, 0.f, 0.f, 0.f, 0.f, 0.f};
    v8f o1 = o0;
    for (int kc = 0; kc < Skv; kc += 32) {
        v16bf aa = load_fragA((const bf16_t*)(s_at + (size_t)(w * 16 + l16) * SP + kc) + half * 16);
        v16bf b0 = load_fragB((const bf16_t*)(s_vt + (size_t)(l16)      * SP + kc) + half * 16);
        v16bf b1 = load_fragB((const bf16_t*)(s_vt + (size_t)(16 + l16) * SP + kc) + half * 16);
        o0 = wmma_bf16(aa, b0, o0);
        o1 = wmma_bf16(aa, b1, o1);
    }
#pragma unroll
    for (int r = 0; r < 8; r++) {
        int m = w * 16 + r + half * 8;
        ((unsigned short*)Ob)[(size_t)m * ors + l16]      = f2bf_bits(o0[r]);
        ((unsigned short*)Ob)[(size_t)m * ors + 16 + l16] = f2bf_bits(o1[r]);
    }
}

// ---------------------------------------------------------------------------
// Host orchestration
// ---------------------------------------------------------------------------
extern "C" void kernel_launch(void* const* d_in, const int* in_sizes, int n_in,
                              void* d_out, int out_size, void* d_ws, size_t ws_size,
                              hipStream_t stream) {
    (void)in_sizes; (void)n_in; (void)out_size; (void)ws_size;
    const float* x_in   = (const float*)d_in[0];
    const float* graph  = (const float*)d_in[1];
    const float* lnf_g  = (const float*)d_in[4];
    const float* lnf_b  = (const float*)d_in[5];
    const float* wqkv_f = (const float*)d_in[6];
    const float* bqkv_f = (const float*)d_in[7];
    const float* wo_f   = (const float*)d_in[8];
    const float* bo_f   = (const float*)d_in[9];
    const float* gb_sc  = (const float*)d_in[10];
    const float* gb_of  = (const float*)d_in[11];
    const float* lntr_g = (const float*)d_in[12];
    const float* lntr_b = (const float*)d_in[13];
    const float* wqkv_tr= (const float*)d_in[14];
    const float* bqkv_tr= (const float*)d_in[15];
    const float* wo_tr  = (const float*)d_in[16];
    const float* bo_tr  = (const float*)d_in[17];
    const float* lnte_g = (const float*)d_in[18];
    const float* lnte_b = (const float*)d_in[19];
    const float* wqkv_te= (const float*)d_in[20];
    const float* bqkv_te= (const float*)d_in[21];
    const float* wo_te  = (const float*)d_in[22];
    const float* bo_te  = (const float*)d_in[23];
    const float* lnff_g = (const float*)d_in[24];
    const float* lnff_b = (const float*)d_in[25];
    const float* w1     = (const float*)d_in[26];
    const float* b1     = (const float*)d_in[27];
    const float* w2     = (const float*)d_in[28];
    const float* b2     = (const float*)d_in[29];

    const int R = 65536;                 // B*S*F rows
    char* ws = (char*)d_ws;
    // arena
    float*  xc   = (float*)(ws);                              // f32  [R,256]  (B,F,S,D)
    bf16_t* nf   = (bf16_t*)(ws + 67108864);                  // bf16 [R,256]
    bf16_t* qkv  = (bf16_t*)(ws + 100663296);                 // bf16 [R,768] (also FFN hidden [R,512])
    bf16_t* att  = (bf16_t*)(ws + 201326592);                 // bf16 [R,256]
    bf16_t* qte  = (bf16_t*)(ws + 234881024);                 // bf16 [16384,256]
    bf16_t* kvte = (bf16_t*)(ws + 243269632);                 // bf16 [49152,512]
    bf16_t* wb   = (bf16_t*)(ws + 293601280);                 // weights bf16
    bf16_t* wb_qkvf = wb;                 // 768*256
    bf16_t* wb_wof  = wb + 196608;        // 256*256
    bf16_t* wb_qkvtr= wb + 262144;        // 768*256
    bf16_t* wb_wotr = wb + 458752;        // 256*256
    bf16_t* wb_qkvte= wb + 524288;        // 768*256
    bf16_t* wb_wote = wb + 720896;        // 256*256
    bf16_t* wb_w1   = wb + 786432;        // 512*256
    bf16_t* wb_w2   = wb + 917504;        // 256*512

    const MapP mp0 = {0, 0};
    const MapP mpSF = {256, 64};   // (b,s,f)->(b,f,s)
    const MapP mpFS = {64, 256};   // (b,f,s)->(b,s,f)

    // --- weights -> bf16
    cvt_bf16_k<<<768, 256, 0, stream>>>(wqkv_f,  wb_qkvf, 196608);
    cvt_bf16_k<<<256, 256, 0, stream>>>(wo_f,    wb_wof,   65536);
    cvt_bf16_k<<<768, 256, 0, stream>>>(wqkv_tr, wb_qkvtr,196608);
    cvt_bf16_k<<<256, 256, 0, stream>>>(wo_tr,   wb_wotr,  65536);
    cvt_bf16_k<<<768, 256, 0, stream>>>(wqkv_te, wb_qkvte,196608);
    cvt_bf16_k<<<256, 256, 0, stream>>>(wo_te,   wb_wote,  65536);
    cvt_bf16_k<<<512, 256, 0, stream>>>(w1,      wb_w1,   131072);
    cvt_bf16_k<<<512, 256, 0, stream>>>(w2,      wb_w2,   131072);

    // ---------------- Stage 1: feature attention -----------------------
    ln_k<0><<<R / 8, 256, 0, stream>>>(x_in, nf, lnf_g, lnf_b, R, mp0);
    gemm_k<0,0,0><<<dim3(12, 512), 256, 0, stream>>>(nf, 256, wb_qkvf, 256,
        bqkv_f, nullptr, 0, qkv, 768, R, 768, mp0, mp0);
    {   // (b,s) pairs: 1024, heads 8, Sq=Skv=64, with graph bias
        dim3 g(1024, 8);
        attn_k<true><<<g, 128, 0, stream>>>(
            qkv,       (long long)64 * 768, 768,
            qkv + 256, (long long)64 * 768, 768,
            qkv + 512, (long long)64 * 768, 768,
            att,       (long long)64 * 256, 256,
            1, 64, graph, gb_sc, gb_of, 256);
    }
    // x = x_in + att @ wo_f^T + bo_f, written transposed into xc (B,F,S,D)
    gemm_k<0,3,1><<<dim3(4, 512), 256, 0, stream>>>(att, 256, wb_wof, 256,
        bo_f, x_in, 256, xc, 256, R, 256, mp0, mpSF);

    // ---------------- Stage 2: train self-attention --------------------
    ln_k<1><<<49152 / 8, 256, 0, stream>>>(xc, nf, lntr_g, lntr_b, 49152, mp0);
    gemm_k<1,1,0><<<dim3(12, 384), 256, 0, stream>>>(nf, 256, wb_qkvtr, 256,
        bqkv_tr, nullptr, 0, qkv, 768, 49152, 768, mp0, mp0);
    {   // (b,f) pairs: 256, heads 8, Sq=Skv=192 (3 q-tiles)
        dim3 g(256 * 3, 8);
        attn_k<false><<<g, 128, 0, stream>>>(
            qkv,       (long long)256 * 768, 768,
            qkv + 256, (long long)256 * 768, 768,
            qkv + 512, (long long)256 * 768, 768,
            att,       (long long)256 * 256, 256,
            3, 192, nullptr, nullptr, nullptr, 1);
    }
    gemm_k<1,1,1><<<dim3(4, 384), 256, 0, stream>>>(att, 256, wb_wotr, 256,
        bo_tr, xc, 256, xc, 256, 49152, 256, mp0, mp0);

    // ---------------- Stage 3: test cross-attention ---------------------
    ln_k<2><<<16384 / 8, 256, 0, stream>>>(xc, nf, lnte_g, lnte_b, 16384, mp0); // te_n
    ln_k<1><<<49152 / 8, 256, 0, stream>>>(xc, nf, lnte_g, lnte_b, 49152, mp0); // kv_n
    gemm_k<2,0,0><<<dim3(4, 128), 256, 0, stream>>>(nf, 256, wb_qkvte, 256,
        bqkv_te, nullptr, 0, qte, 256, 16384, 256, mp0, mp0);                   // Q only
    gemm_k<1,0,0><<<dim3(8, 384), 256, 0, stream>>>(nf, 256, wb_qkvte + 256 * 256, 256,
        bqkv_te + 256, nullptr, 0, kvte, 512, 49152, 512, mp0, mp0);            // K,V
    {   // (b,f) pairs: 256, heads 8, Sq=64, Skv=192
        dim3 g(256, 8);
        attn_k<false><<<g, 128, 0, stream>>>(
            qte,        (long long)64 * 256,  256,
            kvte,       (long long)192 * 512, 512,
            kvte + 256, (long long)192 * 512, 512,
            att + 192 * 256, (long long)256 * 256, 256,
            1, 192, nullptr, nullptr, nullptr, 1);
    }
    gemm_k<2,2,1><<<dim3(4, 128), 256, 0, stream>>>(att, 256, wb_wote, 256,
        bo_te, xc, 256, xc, 256, 16384, 256, mp0, mp0);

    // ---------------- Stage 4: FFN --------------------------------------
    ln_k<0><<<R / 8, 256, 0, stream>>>(xc, nf, lnff_g, lnff_b, R, mp0);
    gemm_k<0,0,2><<<dim3(8, 512), 256, 0, stream>>>(nf, 256, wb_w1, 256,
        b1, nullptr, 0, qkv, 512, R, 512, mp0, mp0);                 // GELU hidden
    // out = xc + h @ w2^T + b2, written back transposed to (B,S,F,D)
    gemm_k<0,3,1><<<dim3(4, 512), 256, 0, stream>>>(qkv, 512, wb_w2, 512,
        b2, xc, 256, (float*)d_out, 256, R, 256, mp0, mpFS);
}
